// SupConLoss_10771777978698
// MI455X (gfx1250) — matrix operand order repaired and checked by reading, not compiled, not run on hardware
//
#include <hip/hip_runtime.h>
#include <hip/hip_bf16.h>

typedef __attribute__((ext_vector_type(16))) _Float16      v16h;
typedef __attribute__((ext_vector_type(8)))  float         v8f;
typedef __attribute__((ext_vector_type(4)))  unsigned int  v4u;
typedef __attribute__((ext_vector_type(8)))  int           v8i;
typedef __attribute__((ext_vector_type(4)))  int           v4i;

#define B_ROWS 8192
#define DIM    1024
#define KSTEPS (DIM / 32)          // 32 WMMA k-steps of 32
#define NSPLIT 8                   // column-range splits (grid.y)
#define WG_ROWS 128                // 8 waves x 16 rows
#define JTILES_PER_SPLIT (B_ROWS / 16 / NSPLIT)  // 64
#define TILE_HALVES (16 * DIM)     // 16384 f16 elements = 32 KB per tile
#define TEMP  0.5f
#define INV_T 2.0f

union Frag { v16h v; uint4 u[2]; };

// ---------------------------------------------------------------------------
// Kernel 1: L2-normalize rows (fp32 math), emit F16 matrix for WMMA.
// ---------------------------------------------------------------------------
__global__ __launch_bounds__(256, 1)
void supcon_normalize(const float* __restrict__ x, _Float16* __restrict__ y)
{
    const int wave = threadIdx.x >> 5;
    const int lane = threadIdx.x & 31;
    const int row  = blockIdx.x * 8 + wave;
    const float4* xr = (const float4*)(x + (size_t)row * DIM);

    float ss = 0.f;
    #pragma unroll
    for (int k = 0; k < DIM / 128; ++k) {
        float4 v = xr[lane + k * 32];
        ss += v.x * v.x + v.y * v.y + v.z * v.z + v.w * v.w;
    }
    #pragma unroll
    for (int off = 16; off > 0; off >>= 1)
        ss += __shfl_xor(ss, off, 32);
    const float inv = 1.0f / fmaxf(sqrtf(ss), 1e-12f);

    _Float16* yr = y + (size_t)row * DIM;
    #pragma unroll
    for (int k = 0; k < DIM / 128; ++k) {
        float4 v = xr[lane + k * 32];
        int base = (lane + k * 32) * 4;
        yr[base + 0] = (_Float16)(v.x * inv);
        yr[base + 1] = (_Float16)(v.y * inv);
        yr[base + 2] = (_Float16)(v.z * inv);
        yr[base + 3] = (_Float16)(v.w * inv);
    }
}

// ---------------------------------------------------------------------------
// TDM: issue a 1-D 32 KB DMA of 16 rows of the F16 matrix into LDS.
// D# packing per CDNA5 ISA §8.3-8.6 (2-D descriptor, groups 2/3 zero).
// 6-arg toolchain form: (g0, g1, g2, g3, g4, cpol).
// ---------------------------------------------------------------------------
__device__ __forceinline__
void tdm_load_tile(const _Float16* gsrc, void* lds_dst)
{
    const unsigned long long ga = (unsigned long long)(uintptr_t)gsrc;
    const unsigned lds = (unsigned)(uintptr_t)lds_dst;  // flat LDS: offset in [31:0]

    v4u g0;
    g0.x = 1u;                                          // count=1 (valid), user mode
    g0.y = lds;                                         // lds_addr [63:32]
    g0.z = (unsigned)(ga & 0xFFFFFFFFu);                // global_addr [95:64]
    g0.w = (unsigned)((ga >> 32) & 0x01FFFFFFu)         // global_addr [120:96]
         | (2u << 30);                                  // type=2 (image) [127:126]

    v8i g1;
    g1[0] = 0x10000;               // wg_mask=0, data_size=1 (2 bytes)
    g1[1] = 0;                     // no atomic barrier; tensor_dim0[15:0]<<16 = 0
    g1[2] = 0x80 | (1 << 16);      // tensor_dim0 = 8M (hi16=0x80); tensor_dim1 lo = 1
    g1[3] = (TILE_HALVES << 16);   // tensor_dim1 hi = 0; tile_dim0 = 16384
    g1[4] = 1;                     // tile_dim1 = 1; tile_dim2 = 0
    g1[5] = 0x800000;              // tensor_dim0_stride = 8M (low 32)
    g1[6] = 0;
    g1[7] = 0;

    v4i z4 = {0, 0, 0, 0};
    v8i z8 = {0, 0, 0, 0, 0, 0, 0, 0};
    __builtin_amdgcn_tensor_load_to_lds(g0, g1, z4, z4, z8, 0);
}

// ---------------------------------------------------------------------------
// Kernel 2: fused Gram-matrix (WMMA f32_16x16x32_f16) + SupCon reductions.
// Per wave: 16 rows, A panel (16x1024 f16) resident in 256 VGPRs.
// B tiles double-buffered in LDS, filled by the Tensor Data Mover while
// all 8 waves run the 32-step WMMA chain on the other buffer.
// ---------------------------------------------------------------------------
__global__ __launch_bounds__(256, 1)
void supcon_gram(const _Float16* __restrict__ f,
                 const int* __restrict__ labels,
                 float* __restrict__ pE, float* __restrict__ pP,
                 float* __restrict__ pN)
{
    __shared__ _Float16 Bt[2][TILE_HALVES];   // 2 x 32 KB double buffer

    const int tid  = threadIdx.x;
    const int wave = tid >> 5;
    const int lane = tid & 31;
    const int hi   = lane >> 4;         // which 16-lane half
    const int ln   = lane & 15;

    const int rowBase = blockIdx.x * WG_ROWS + wave * 16;
    const int jStart  = blockIdx.y * (B_ROWS / NSPLIT);

    // A fragments: lane ln holds row rowBase+ln; half 'hi' selects the
    // interleaved K chunks {k0+8h..+8, k0+16+8h..+8} per the 16-bit A layout.
    Frag a[KSTEPS];
    {
        const _Float16* ar = f + (size_t)(rowBase + ln) * DIM;
        #pragma unroll
        for (int ks = 0; ks < KSTEPS; ++ks) {
            const int k0 = ks * 32 + hi * 8;
            a[ks].u[0] = *(const uint4*)(ar + k0);
            a[ks].u[1] = *(const uint4*)(ar + k0 + 16);
        }
    }

    // Labels of the 8 output rows this lane's accumulator covers (M = v + 8*hi).
    int rowLab[8];
    #pragma unroll
    for (int v = 0; v < 8; ++v)
        rowLab[v] = labels[rowBase + v + 8 * hi];

    float accE[8], accP[8], accN[8];
    #pragma unroll
    for (int v = 0; v < 8; ++v) { accE[v] = 0.f; accP[v] = 0.f; accN[v] = 0.f; }

    // Prologue: TDM-fill buffer 0 with the first tile.
    if (wave == 0) {
        tdm_load_tile(f + (size_t)jStart * DIM, &Bt[0][0]);
        __builtin_amdgcn_s_wait_tensorcnt(0);
    }
    __syncthreads();

    int buf = 0;
    for (int jt = 0; jt < JTILES_PER_SPLIT; ++jt) {
        const int jbase = jStart + jt * 16;

        // Kick off the next tile's DMA into the other buffer (overlaps compute).
        if (wave == 0 && jt + 1 < JTILES_PER_SPLIT)
            tdm_load_tile(f + (size_t)(jbase + 16) * DIM, &Bt[buf ^ 1][0]);

        // 16x16 f32 accumulator over K = 1024
        v8f c = {0.f, 0.f, 0.f, 0.f, 0.f, 0.f, 0.f, 0.f};
        const _Float16* br = &Bt[buf][0] + (size_t)ln * DIM + hi * 8;
        #pragma unroll
        for (int ks = 0; ks < KSTEPS; ++ks) {
            Frag b;
            b.u[0] = *(const uint4*)(br + ks * 32);
            b.u[1] = *(const uint4*)(br + ks * 32 + 16);
            c = __builtin_amdgcn_wmma_f32_16x16x32_f16(
                    false, a[ks].v, false, b.v, (short)0, c, false, false);
        }

        // Epilogue: logits = s/T, analytic row max = 1/T -> shifted = (s-1)/T.
        const int j    = jbase + ln;      // this lane's column index
        const int labj = labels[j];
        #pragma unroll
        for (int v = 0; v < 8; ++v) {
            const int rowv   = rowBase + v + 8 * hi;
            const float sh   = (c[v] - 1.0f) * INV_T;
            const bool self  = (j == rowv);
            const bool pos   = (!self) && (labj == rowLab[v]);
            accE[v] += self ? 0.f : __expf(sh);
            accP[v] += pos ? sh  : 0.f;
            accN[v] += pos ? 1.f : 0.f;
        }

        // Make the incoming tile visible, flip buffers.
        if (wave == 0)
            __builtin_amdgcn_s_wait_tensorcnt(0);
        __syncthreads();
        buf ^= 1;
    }

    // Reduce the 16 lane-columns of each half; lanes 0 / 16 own rows v / v+8.
    #pragma unroll
    for (int v = 0; v < 8; ++v) {
        float e = accE[v], p = accP[v], n = accN[v];
        #pragma unroll
        for (int off = 8; off > 0; off >>= 1) {
            e += __shfl_xor(e, off, 32);
            p += __shfl_xor(p, off, 32);
            n += __shfl_xor(n, off, 32);
        }
        if (ln == 0) {
            const int row = rowBase + v + 8 * hi;
            const size_t idx = (size_t)blockIdx.y * B_ROWS + row;
            pE[idx] = e; pP[idx] = p; pN[idx] = n;
        }
    }
}

// ---------------------------------------------------------------------------
// Kernel 3: combine splits, per-row loss, mean over valid rows -> scalar.
// ---------------------------------------------------------------------------
__global__ __launch_bounds__(256)
void supcon_finalize(const float* __restrict__ pE, const float* __restrict__ pP,
                     const float* __restrict__ pN, float* __restrict__ out)
{
    __shared__ float sL[256], sC[256];
    float sumL = 0.f, cnt = 0.f;
    for (int r = threadIdx.x; r < B_ROWS; r += 256) {
        float E = 0.f, P = 0.f, N = 0.f;
        #pragma unroll
        for (int s = 0; s < NSPLIT; ++s) {
            const size_t idx = (size_t)s * B_ROWS + r;
            E += pE[idx]; P += pP[idx]; N += pN[idx];
        }
        if (N > 0.f) {
            sumL += -(P / N - __logf(E)) * TEMP;   // loss_i = -(P/N - log E)*T
            cnt  += 1.f;
        }
    }
    sL[threadIdx.x] = sumL; sC[threadIdx.x] = cnt;
    __syncthreads();
    for (int s = 128; s > 0; s >>= 1) {
        if (threadIdx.x < s) {
            sL[threadIdx.x] += sL[threadIdx.x + s];
            sC[threadIdx.x] += sC[threadIdx.x + s];
        }
        __syncthreads();
    }
    if (threadIdx.x == 0) out[0] = sL[0] / fmaxf(sC[0], 1.f);
}

// ---------------------------------------------------------------------------
extern "C" void kernel_launch(void* const* d_in, const int* in_sizes, int n_in,
                              void* d_out, int out_size, void* d_ws, size_t ws_size,
                              hipStream_t stream)
{
    const float* feats  = (const float*)d_in[0];
    const int*   labels = (const int*)d_in[1];

    _Float16* fnorm = (_Float16*)d_ws;                                   // 16 MB
    float* pE = (float*)((char*)d_ws + (size_t)B_ROWS * DIM * sizeof(_Float16));
    float* pP = pE + (size_t)NSPLIT * B_ROWS;
    float* pN = pP + (size_t)NSPLIT * B_ROWS;

    supcon_normalize<<<B_ROWS / 8, 256, 0, stream>>>(feats, fnorm);

    dim3 grid(B_ROWS / WG_ROWS, NSPLIT);
    supcon_gram<<<grid, 256, 0, stream>>>(fnorm, labels, pE, pP, pN);

    supcon_finalize<<<1, 256, 0, stream>>>(pE, pP, pN, (float*)d_out);
}
